// GraphRec_46076409152415
// MI455X (gfx1250) — matrix operand order repaired and checked by reading, not compiled, not run on hardware
//
#include <hip/hip_runtime.h>

typedef _Float16 h16;
typedef __attribute__((ext_vector_type(16))) _Float16 v16h;
typedef __attribute__((ext_vector_type(8)))  _Float16 v8h;
typedef __attribute__((ext_vector_type(8)))  float    v8f;

// ---------------------------------------------------------------------------
// f32 -> f16 conversion WITH swizzle into WMMA B-fragment order.
// src is row-major (K x 64). dst layout: tile t = kt*4 + n  (kt = K-tile of 32,
// n = N-tile of 16); within a tile, 32 lanes x 16 contiguous halves where
// element i of lane L is  src[(kt*32 + (i>>3)*16 + (L>>4)*8 + (i&7))*64 + n*16 + (L&15)].
// This makes the per-lane B-fragment a single contiguous 32-byte LDS read.
// ---------------------------------------------------------------------------
__global__ void cvt_swizzle_kernel(const float* __restrict__ src,
                                   h16* __restrict__ dst, int K) {
  int i = blockIdx.x * 256 + threadIdx.x;
  if (i >= K * 64) return;
  int elem = i & 15;
  int lane = (i >> 4) & 31;
  int tile = i >> 9;                 // = kt*4 + n
  int kt = tile >> 2, n = tile & 3;
  int krow = kt * 32 + ((elem >> 3) << 4) + ((lane >> 4) << 3) + (elem & 7);
  int col  = (n << 4) + (lane & 15);
  dst[i] = (h16)src[krow * 64 + col];
}

// ---------------------------------------------------------------------------
// Workgroup GEMM: (64 x K) @ (K x 64) + bias (+relu), f16 in / f32 acc / f16 out
// 128 threads = 4 waves; wave w owns M-rows [16w,16w+16).
// A: row-major in LDS (two contiguous v8h chunks per lane per fragment).
// W: pre-swizzled B-fragment order (one contiguous v16h per lane per fragment).
// ---------------------------------------------------------------------------
template <int K>
__device__ __forceinline__ void wg_gemm(const h16* __restrict__ A, int a_stride,
                                        const h16* __restrict__ W,
                                        const float* __restrict__ bias_s,
                                        bool do_relu,
                                        h16* __restrict__ Out, int o_stride,
                                        int tid) {
  const int wave  = tid >> 5;
  const int lane  = tid & 31;
  const int r0    = wave << 4;
  const int halfg = lane >> 4;   // lane group 0: lanes 0-15, 1: lanes 16-31
  const int l15   = lane & 15;
#pragma unroll
  for (int n = 0; n < 4; ++n) {
    v8f c = {};
#pragma unroll
    for (int kk = 0; kk < K / 32; ++kk) {
      const int k0 = kk * 32;
      // A fragment 16x32: halves 0..7 -> K = 8*halfg + i,
      // halves 8..15 -> K = 16 + 8*halfg + i  (two contiguous 16B chunks)
      const h16* abase = A + (r0 + l15) * a_stride + k0 + (halfg << 3);
      v8h alo = *(const v8h*)(abase);
      v8h ahi = *(const v8h*)(abase + 16);
      v16h a = __builtin_shufflevector(alo, ahi, 0, 1, 2, 3, 4, 5, 6, 7,
                                       8, 9, 10, 11, 12, 13, 14, 15);
      // B fragment: contiguous 32 bytes per lane (pre-swizzled)
      const h16* bbase = W + ((((kk << 2) + n) << 5) + lane) * 16;
      v16h b = *(const v16h*)bbase;
      c = __builtin_amdgcn_wmma_f32_16x16x32_f16(false, a, false, b,
                                                 (short)0, c, false, false);
    }
    const int col = (n << 4) + l15;
    const float bia = bias_s[col];
#pragma unroll
    for (int j = 0; j < 8; ++j) {
      const int row = r0 + j + (halfg << 3);   // C/D: vgpr j -> M = j + 8*halfg
      float v = c[j] + bia;
      if (do_relu) v = fmaxf(v, 0.0f);
      Out[row * o_stride + col] = (h16)v;
    }
  }
}

// 16-byte chunked LDS staging of a pre-swizzled weight (K*64 halves)
__device__ __forceinline__ void stage_w(const h16* __restrict__ g,
                                        h16* __restrict__ s, int Khalves,
                                        int tid) {
  const uint4* gs = (const uint4*)g;
  uint4* ss = (uint4*)s;
  for (int i = tid; i < Khalves / 8; i += 128) ss[i] = gs[i];
}

// ---------------------------------------------------------------------------
// Fused neighbor aggregation (one entity per workgroup, 128 threads).
// mode 0: item-space agg for user `blockIdx.x` (ia weights, C/R tables)
// mode 1: social agg for batch elem (sa weights, x = h_I_all[N[u]])
// mode 2: user-space agg for batch item (ua weights, B/R tables)
// ---------------------------------------------------------------------------
__global__ __launch_bounds__(128) void agg_kernel(
    int mode, const int* __restrict__ batch_idx,
    const int* __restrict__ nbr_table, const int* __restrict__ R,
    const float* __restrict__ emb, const float* __restrict__ opin,
    const float* __restrict__ ctx_emb,
    const h16* __restrict__ Wg16, const float* __restrict__ bg,
    const h16* __restrict__ Wa1_16, const float* __restrict__ ba1,
    const float* __restrict__ Wa2, const float* __restrict__ ba2,
    const float* __restrict__ W, const float* __restrict__ bvec,
    float* __restrict__ out) {
  __shared__ alignas(16) h16 Ebuf[64 * 128];   // gate input [emb ; opinion]
  __shared__ alignas(16) h16 Xbuf[64 * 128];   // [x ; ctx] attention input
  __shared__ alignas(16) h16 Wbuf[128 * 64];   // staged weight (Wg then Wa1)
  __shared__ alignas(16) h16 Hbuf[64 * 64];    // attention hidden
  __shared__ float ctxf[64], biasbuf[64], scores[64], avec[64], hx[64];
  __shared__ int nbr[64], rat[64];

  const int e = blockIdx.x;
  const int tid = threadIdx.x;
  const int ent = (mode == 0) ? e : batch_idx[e];

  if (tid < 64) ctxf[tid] = ctx_emb[ent * 64 + tid];
  if (tid >= 64) {
    int k = tid - 64, id = 0, rr = 0;
    if (k < 50) {
      id = nbr_table[ent * 50 + k];
      if (mode == 0) rr = R[ent * 10000 + id];
      else if (mode == 2) rr = R[id * 10000 + ent];
    }
    nbr[k] = id; rat[k] = rr;
  }
  __syncthreads();

  // ctx half of attention input (same for every row; masked rows die in softmax)
  for (int i = tid; i < 64 * 64; i += 128) {
    int r = i >> 6, c = i & 63;
    Xbuf[r * 128 + 64 + c] = (h16)ctxf[c];
  }
  if (mode == 1) {
    for (int i = tid; i < 64 * 64; i += 128) {
      int r = i >> 6, c = i & 63;
      float v = (r < 50) ? emb[nbr[r] * 64 + c] : 0.0f;
      Xbuf[r * 128 + c] = (h16)v;
    }
  } else {
    for (int i = tid; i < 64 * 128; i += 128) {
      int r = i >> 7, c = i & 127;
      float v = 0.0f;
      if (r < 50)
        v = (c < 64) ? emb[nbr[r] * 64 + c] : opin[rat[r] * 64 + (c - 64)];
      Ebuf[r * 128 + c] = (h16)v;
    }
    stage_w(Wg16, Wbuf, 128 * 64, tid);
    if (tid < 64) biasbuf[tid] = bg[tid];
  }
  __syncthreads();

  if (mode != 1) {  // gate GEMM: x = relu([emb;opin] @ Wg + bg) -> Xbuf[:, :64]
    wg_gemm<128>(Ebuf, 128, Wbuf, biasbuf, true, Xbuf, 128, tid);
    __syncthreads();
  }

  // attention hidden: h = relu([x;ctx] @ Wa1 + ba1)
  stage_w(Wa1_16, Wbuf, 128 * 64, tid);
  if (tid < 64) biasbuf[tid] = ba1[tid];
  __syncthreads();
  wg_gemm<128>(Xbuf, 128, Wbuf, biasbuf, true, Hbuf, 64, tid);
  __syncthreads();

  // scores = h @ Wa2 + ba2
  if (tid < 64) {
    float s = ba2[0];
    for (int d = 0; d < 64; ++d) s += (float)Hbuf[tid * 64 + d] * Wa2[d];
    scores[tid] = s;
  }
  __syncthreads();
  if (tid == 0) {
    float m = -1e30f;
    for (int k = 0; k < 50; ++k) m = fmaxf(m, scores[k]);
    float ssum = 0.0f;
    for (int k = 0; k < 50; ++k) { float v = __expf(scores[k] - m); avec[k] = v; ssum += v; }
    float inv = 1.0f / ssum;
    for (int k = 0; k < 50; ++k) avec[k] *= inv;
    for (int k = 50; k < 64; ++k) avec[k] = 0.0f;
  }
  __syncthreads();
  // weighted sum over neighbors
  if (tid < 64) {
    float acc = 0.0f;
    for (int k = 0; k < 50; ++k) acc += avec[k] * (float)Xbuf[k * 128 + tid];
    hx[tid] = acc;
  }
  __syncthreads();
  // output projection: relu(hx @ W + b), fp32
  if (tid < 64) {
    float acc = bvec[tid];
    for (int d = 0; d < 64; ++d) acc += hx[d] * W[d * 64 + tid];
    out[e * 64 + tid] = fmaxf(acc, 0.0f);
  }
}

// ---------------------------------------------------------------------------
// Fusion + rating-prediction MLP head: 64 batch rows per workgroup.
// ---------------------------------------------------------------------------
__global__ __launch_bounds__(128) void mlp_head_kernel(
    const int* __restrict__ user_idx, const float* __restrict__ h_I_all,
    const float* __restrict__ h_S, const float* __restrict__ z,
    const h16* __restrict__ fuW1, const float* __restrict__ fub1,
    const h16* __restrict__ fuW2, const float* __restrict__ fub2,
    const h16* __restrict__ fuW3, const float* __restrict__ fub3,
    const h16* __restrict__ rpW1, const float* __restrict__ rpb1,
    const h16* __restrict__ rpW2, const float* __restrict__ rpb2,
    const float* __restrict__ rpW3, const float* __restrict__ rpb3,
    float* __restrict__ out) {
  __shared__ alignas(16) h16 Buf128[64 * 128];
  __shared__ alignas(16) h16 BufA[64 * 64];
  __shared__ alignas(16) h16 BufB[64 * 64];
  __shared__ alignas(16) h16 Wbuf[128 * 64];
  __shared__ float biasbuf[64];

  const int b0 = blockIdx.x * 64;
  const int tid = threadIdx.x;

  for (int i = tid; i < 64 * 64; i += 128) {
    int r = i >> 6, c = i & 63;
    int u = user_idx[b0 + r];
    Buf128[r * 128 + c]      = (h16)h_I_all[u * 64 + c];        // h_I_batch
    Buf128[r * 128 + 64 + c] = (h16)h_S[(b0 + r) * 64 + c];     // h_S
  }
  stage_w(fuW1, Wbuf, 128 * 64, tid);
  if (tid < 64) biasbuf[tid] = fub1[tid];
  __syncthreads();
  wg_gemm<128>(Buf128, 128, Wbuf, biasbuf, true, BufA, 64, tid);
  __syncthreads();

  stage_w(fuW2, Wbuf, 64 * 64, tid);
  if (tid < 64) biasbuf[tid] = fub2[tid];
  __syncthreads();
  wg_gemm<64>(BufA, 64, Wbuf, biasbuf, true, BufB, 64, tid);
  __syncthreads();

  stage_w(fuW3, Wbuf, 64 * 64, tid);
  if (tid < 64) biasbuf[tid] = fub3[tid];
  __syncthreads();
  wg_gemm<64>(BufB, 64, Wbuf, biasbuf, true, Buf128, 128, tid);  // h -> cols 0-63
  for (int i = tid; i < 64 * 64; i += 128) {
    int r = i >> 6, c = i & 63;
    Buf128[r * 128 + 64 + c] = (h16)z[(b0 + r) * 64 + c];        // g = [h ; z]
  }
  __syncthreads();

  stage_w(rpW1, Wbuf, 128 * 64, tid);
  if (tid < 64) biasbuf[tid] = rpb1[tid];
  __syncthreads();
  wg_gemm<128>(Buf128, 128, Wbuf, biasbuf, true, BufA, 64, tid);
  __syncthreads();

  stage_w(rpW2, Wbuf, 64 * 64, tid);
  if (tid < 64) biasbuf[tid] = rpb2[tid];
  __syncthreads();
  wg_gemm<64>(BufA, 64, Wbuf, biasbuf, true, BufB, 64, tid);
  __syncthreads();

  if (tid < 64) {  // final 64 -> 1 projection (no relu)
    float acc = rpb3[0];
    for (int d = 0; d < 64; ++d) acc += (float)BufB[tid * 64 + d] * rpW3[d];
    out[b0 + tid] = acc;
  }
}

// ---------------------------------------------------------------------------
extern "C" void kernel_launch(void* const* d_in, const int* in_sizes, int n_in,
                              void* d_out, int out_size, void* d_ws,
                              size_t ws_size, hipStream_t stream) {
  (void)in_sizes; (void)n_in; (void)out_size; (void)ws_size;
  const int*   user_idx  = (const int*)d_in[0];
  const int*   item_idx  = (const int*)d_in[1];
  const int*   C         = (const int*)d_in[2];
  const int*   N         = (const int*)d_in[3];
  const int*   B         = (const int*)d_in[4];
  const int*   R         = (const int*)d_in[5];
  const float* user_emb  = (const float*)d_in[6];
  const float* item_emb  = (const float*)d_in[7];
  const float* opinion   = (const float*)d_in[8];
  const float* ia_Wg  = (const float*)d_in[9];  const float* ia_bg  = (const float*)d_in[10];
  const float* ua_Wg  = (const float*)d_in[11]; const float* ua_bg  = (const float*)d_in[12];
  const float* ia_Wa1 = (const float*)d_in[13]; const float* ia_ba1 = (const float*)d_in[14];
  const float* ia_Wa2 = (const float*)d_in[15]; const float* ia_ba2 = (const float*)d_in[16];
  const float* ia_W   = (const float*)d_in[17]; const float* ia_b   = (const float*)d_in[18];
  const float* sa_Wa1 = (const float*)d_in[19]; const float* sa_ba1 = (const float*)d_in[20];
  const float* sa_Wa2 = (const float*)d_in[21]; const float* sa_ba2 = (const float*)d_in[22];
  const float* sa_W   = (const float*)d_in[23]; const float* sa_b   = (const float*)d_in[24];
  const float* ua_Wa1 = (const float*)d_in[25]; const float* ua_ba1 = (const float*)d_in[26];
  const float* ua_Wa2 = (const float*)d_in[27]; const float* ua_ba2 = (const float*)d_in[28];
  const float* ua_W   = (const float*)d_in[29]; const float* ua_b   = (const float*)d_in[30];
  const float* fu_W1  = (const float*)d_in[31]; const float* fu_b1  = (const float*)d_in[32];
  const float* fu_W2  = (const float*)d_in[33]; const float* fu_b2  = (const float*)d_in[34];
  const float* rp_W1  = (const float*)d_in[35]; const float* rp_b1  = (const float*)d_in[36];
  const float* rp_W2  = (const float*)d_in[37]; const float* rp_b2  = (const float*)d_in[38];
  const float* fu_W3  = (const float*)d_in[39]; const float* fu_b3  = (const float*)d_in[40];
  const float* rp_W3  = (const float*)d_in[41]; const float* rp_b3  = (const float*)d_in[42];

  char* ws = (char*)d_ws;
  float* h_I_all = (float*)ws;                               // 10000*64 f32
  float* h_S     = (float*)(ws + 2560000);                   // 2048*64 f32
  float* zbuf    = (float*)(ws + 2560000 + 524288);          // 2048*64 f32
  h16*   w16     = (h16*)(ws + 2560000 + 2 * 524288);
  h16* iaWg16  = w16;          h16* iaWa1_16 = w16 + 8192;
  h16* saWa1_16 = w16 + 16384; h16* uaWg16   = w16 + 24576;
  h16* uaWa1_16 = w16 + 32768; h16* fuW1_16  = w16 + 40960;
  h16* fuW2_16  = w16 + 49152; h16* fuW3_16  = w16 + 53248;
  h16* rpW1_16  = w16 + 57344; h16* rpW2_16  = w16 + 65536;

  auto cvt = [&](const float* s, h16* d, int K) {
    cvt_swizzle_kernel<<<(K * 64 + 255) / 256, 256, 0, stream>>>(s, d, K);
  };
  cvt(ia_Wg, iaWg16, 128);   cvt(ia_Wa1, iaWa1_16, 128);
  cvt(sa_Wa1, saWa1_16, 128);cvt(ua_Wg, uaWg16, 128);
  cvt(ua_Wa1, uaWa1_16, 128);cvt(fu_W1, fuW1_16, 128);
  cvt(fu_W2, fuW2_16, 64);   cvt(fu_W3, fuW3_16, 64);
  cvt(rp_W1, rpW1_16, 128);  cvt(rp_W2, rpW2_16, 64);

  // h_I_all for all 10000 users (h_I_batch is a gather of this)
  agg_kernel<<<10000, 128, 0, stream>>>(
      0, nullptr, C, R, item_emb, opinion, user_emb,
      iaWg16, ia_bg, iaWa1_16, ia_ba1, ia_Wa2, ia_ba2, ia_W, ia_b, h_I_all);

  // item-side aggregation z (independent of h_I_all)
  agg_kernel<<<2048, 128, 0, stream>>>(
      2, item_idx, B, R, user_emb, opinion, item_emb,
      uaWg16, ua_bg, uaWa1_16, ua_ba1, ua_Wa2, ua_ba2, ua_W, ua_b, zbuf);

  // social aggregation h_S (consumes h_I_all)
  agg_kernel<<<2048, 128, 0, stream>>>(
      1, user_idx, N, nullptr, h_I_all, nullptr, user_emb,
      nullptr, nullptr, saWa1_16, sa_ba1, sa_Wa2, sa_ba2, sa_W, sa_b, h_S);

  // fusion + rating prediction head
  mlp_head_kernel<<<32, 128, 0, stream>>>(
      user_idx, h_I_all, h_S, zbuf,
      fuW1_16, fu_b1, fuW2_16, fu_b2, fuW3_16, fu_b3,
      rpW1_16, rp_b1, rpW2_16, rp_b2, rp_W3, rp_b3, (float*)d_out);
}
